// MultiHeadAttention_90812788506828
// MI455X (gfx1250) — matrix operand order repaired and checked by reading, not compiled, not run on hardware
//
#include <hip/hip_runtime.h>
#include <hip/hip_bf16.h>

#define DIM   768
#define HEADS 12
#define NTOK  64
#define HD    64

typedef __bf16 bf16;
typedef __attribute__((ext_vector_type(16))) __bf16 v16bf;
typedef __attribute__((ext_vector_type(8)))  __bf16 v8bf;
typedef __attribute__((ext_vector_type(8)))  float  v8f;

union V16 { v16bf v; v8bf h[2]; };

__device__ __forceinline__ bf16 f2bf(float f) {
    unsigned u = __builtin_bit_cast(unsigned, f);
    u += 0x7FFFu + ((u >> 16) & 1u);               // round-to-nearest-even
    unsigned short hs = (unsigned short)(u >> 16);
    return __builtin_bit_cast(bf16, hs);
}

// CDNA5 async copy: LDS[lds_off(lane)] = MEM[g(lane)], 16B per lane, tracked by ASYNCcnt
__device__ __forceinline__ void async_ld_b128(unsigned lds_off, const void* g) {
    asm volatile("global_load_async_to_lds_b128 %0, %1, off"
                 :: "v"(lds_off), "v"(g) : "memory");
}
__device__ __forceinline__ void wait_async0() {
    asm volatile("s_wait_asynccnt 0" ::: "memory");
}

// ---------------- kernel 0a: fp32 -> bf16 weight conversion ----------------
__global__ __launch_bounds__(256) void convert_f32_bf16(const float* __restrict__ s,
                                                        bf16* __restrict__ d, int n) {
    int i = blockIdx.x * 256 + threadIdx.x;
    if (i < n) d[i] = f2bf(s[i]);
}

// ---------------- kernel 0b: gather relative position bias -> (H,64,64) f32 ----
__global__ __launch_bounds__(256) void bias_gather(const float* __restrict__ tbl,
                                                   const int* __restrict__ idx,
                                                   float* __restrict__ bias) {
    int t  = blockIdx.x * 256 + threadIdx.x;   // t = h*4096 + n*64 + m, t < 49152
    int h  = t >> 12;
    int nm = t & 4095;
    bias[t] = tbl[idx[nm] * HEADS + h];
}

// ---------------- bf16 WMMA GEMM, 128x64 block tile, BK=32, 8 waves ---------
// wave w owns M-strip w (16 rows) x all four 16-col N-subtiles: 4 WMMAs / K-step
// MODE 0: A = fp32 (x), epilogue -> Q (scaled bf16), K (bf16), Vt (bf16 transposed)
// MODE 1: A = bf16 (ctx), epilogue adds bias, writes f32 out
template<int MODE>
__global__ __launch_bounds__(256) void gemm_k(
    const void* __restrict__ Aptr, const bf16* __restrict__ Bw,
    const float* __restrict__ biasN,
    bf16* __restrict__ Qb, bf16* __restrict__ Kb, bf16* __restrict__ Vt,
    float* __restrict__ outF, int M, int K, int N)
{
    __shared__ bf16 As[128 * 40];  // 128 rows (M) x 32 (K), stride 40
    __shared__ bf16 Bs[64 * 40];   // transposed: 64 cols (N) x 32 (K), stride 40

    const int tid    = threadIdx.x;
    const int wave   = tid >> 5;          // 0..7 = M sub-tile
    const int lane   = tid & 31;
    const int lane16 = lane & 15;
    const int hi     = lane >> 4;

    const int m0 = blockIdx.y * 128;
    const int n0 = blockIdx.x * 64;

    const v8f zero = {0.f,0.f,0.f,0.f,0.f,0.f,0.f,0.f};
    v8f acc[4] = {zero, zero, zero, zero};

    // cooperative loader coordinates
    const int arow = tid >> 1;            // 0..127
    const int acol = (tid & 1) * 16;      // 0 or 16
    const int brow = tid >> 3;            // 0..31
    const int bcol = (tid & 7) * 8;       // 0..56

    for (int kt = 0; kt < K; kt += 32) {
        // ---- stage global -> registers (next tile prefetched) ----
        v8bf av0, av1;
        if (MODE == 0) {
            const float* A = (const float*)Aptr;
            const float4* p = (const float4*)(A + (size_t)(m0 + arow) * K + kt + acol);
            float4 f0 = p[0], f1 = p[1], f2 = p[2], f3 = p[3];
            av0[0]=f2bf(f0.x); av0[1]=f2bf(f0.y); av0[2]=f2bf(f0.z); av0[3]=f2bf(f0.w);
            av0[4]=f2bf(f1.x); av0[5]=f2bf(f1.y); av0[6]=f2bf(f1.z); av0[7]=f2bf(f1.w);
            av1[0]=f2bf(f2.x); av1[1]=f2bf(f2.y); av1[2]=f2bf(f2.z); av1[3]=f2bf(f2.w);
            av1[4]=f2bf(f3.x); av1[5]=f2bf(f3.y); av1[6]=f2bf(f3.z); av1[7]=f2bf(f3.w);
            if (kt + 32 < K) __builtin_prefetch(p + 8, 0, 3);
        } else {
            const bf16* A = (const bf16*)Aptr;
            const bf16* ap = A + (size_t)(m0 + arow) * K + kt + acol;
            av0 = *(const v8bf*)ap;
            av1 = *(const v8bf*)(ap + 8);
            if (kt + 32 < K) __builtin_prefetch(ap + 32, 0, 3);
        }
        v8bf bv = *(const v8bf*)(Bw + (size_t)(kt + brow) * N + n0 + bcol);

        __syncthreads();
        *(v8bf*)(As + arow * 40 + acol)     = av0;
        *(v8bf*)(As + arow * 40 + acol + 8) = av1;
        #pragma unroll
        for (int i = 0; i < 8; ++i) Bs[(bcol + i) * 40 + brow] = bv[i];
        __syncthreads();

        // ---- fragments per ISA layout ----
        V16 a;
        const bf16* ap = As + (wave * 16 + lane16) * 40 + (hi ? 8 : 0);
        a.h[0] = *(const v8bf*)ap;
        a.h[1] = *(const v8bf*)(ap + 16);

        #pragma unroll
        for (int nt = 0; nt < 4; ++nt) {
            V16 bf;
            const bf16* bp = Bs + (nt * 16 + lane16) * 40 + (hi ? 8 : 0);
            bf.h[0] = *(const v8bf*)bp;
            bf.h[1] = *(const v8bf*)(bp + 16);
            acc[nt] = __builtin_amdgcn_wmma_f32_16x16x32_bf16(false, a.v, false, bf.v,
                                                              (short)0, acc[nt], false, false);
        }
    }

    // ---- epilogue ----
    #pragma unroll
    for (int nt = 0; nt < 4; ++nt) {
        const int colg = n0 + nt * 16 + lane16;
        const float bb = biasN[colg];
        if (MODE == 0) {
            const int which = colg / DIM;         // 0=q 1=k 2=v
            const int rem   = colg % DIM;
            const int h     = rem >> 6;
            const int d     = rem & 63;
            #pragma unroll
            for (int r = 0; r < 8; ++r) {
                const int rowg = m0 + wave * 16 + r + hi * 8;
                const int b    = rowg >> 6;
                const int tok  = rowg & 63;
                const int bh   = b * HEADS + h;
                float v = acc[nt][r] + bb;
                if (which == 0)      Qb[(size_t)(bh * 64 + tok) * 64 + d] = f2bf(v * 0.125f);
                else if (which == 1) Kb[(size_t)(bh * 64 + tok) * 64 + d] = f2bf(v);
                else                 Vt[(size_t)(bh * 64 + d) * 64 + tok] = f2bf(v);
            }
        } else {
            #pragma unroll
            for (int r = 0; r < 8; ++r) {
                const int rowg = m0 + wave * 16 + r + hi * 8;
                outF[(size_t)rowg * N + colg] = acc[nt][r] + bb;
            }
        }
    }
}

// ---------------- attention: one block per (window, head), 4 waves ----------
// K and Vt tiles (8 KB each) staged once per block into padded LDS via
// global_load_async_to_lds_b128 (ASYNCcnt); Q fragments read directly (no reuse).
__global__ __launch_bounds__(128) void attn_k(
    const bf16* __restrict__ Qb, const bf16* __restrict__ Kb,
    const bf16* __restrict__ Vt, const float* __restrict__ biasHNN,
    bf16* __restrict__ ctx)
{
    __shared__ bf16 Ks[64 * 72];                   // 64 rows x 64, stride 72 (16B row pad)
    __shared__ bf16 Vs[64 * 72];
    __shared__ bf16 Ps[4 * 16 * 72];               // per-wave 16x64 P strip, stride 72

    const int bh   = blockIdx.x;
    const int b    = bh / HEADS;
    const int h    = bh % HEADS;
    const int wave = threadIdx.x >> 5;
    const int lane = threadIdx.x & 31;
    const int lane16 = lane & 15;
    const int hi     = lane >> 4;

    const bf16* Qp = Qb + (size_t)bh * 4096;
    const bf16* Kp = Kb + (size_t)bh * 4096;
    const bf16* Vp = Vt + (size_t)bh * 4096;

    // ---- async stage K and Vt into LDS with padded rows ----
    {
        const unsigned ldsK = (unsigned)(uintptr_t)(&Ks[0]);
        const unsigned ldsV = (unsigned)(uintptr_t)(&Vs[0]);
        #pragma unroll
        for (int i = 0; i < 4; ++i) {
            const int chunk = threadIdx.x + i * 128;   // 0..511 : 16B chunks of 8KB tile
            const int tok = chunk >> 3;
            const int cw  = chunk & 7;
            const unsigned loff = (unsigned)(tok * 144 + cw * 16);
            async_ld_b128(ldsK + loff, Kp + tok * 64 + cw * 8);
            async_ld_b128(ldsV + loff, Vp + tok * 64 + cw * 8);
        }
        wait_async0();
        __syncthreads();
    }

    const v8f zero = {0.f,0.f,0.f,0.f,0.f,0.f,0.f,0.f};
    v8f s[4] = {zero, zero, zero, zero};

    // ---- S = Q @ K^T  (16x64 strip per wave) ----
    #pragma unroll
    for (int k0 = 0; k0 < HD; k0 += 32) {
        V16 a;
        const bf16* ap = Qp + (wave * 16 + lane16) * 64 + k0 + (hi ? 8 : 0);
        a.h[0] = *(const v8bf*)ap;
        a.h[1] = *(const v8bf*)(ap + 16);
        #pragma unroll
        for (int nt = 0; nt < 4; ++nt) {
            V16 bm;
            const bf16* bp = Ks + (nt * 16 + lane16) * 72 + k0 + (hi ? 8 : 0);
            bm.h[0] = *(const v8bf*)bp;
            bm.h[1] = *(const v8bf*)(bp + 16);
            s[nt] = __builtin_amdgcn_wmma_f32_16x16x32_bf16(false, a.v, false, bm.v,
                                                            (short)0, s[nt], false, false);
        }
    }

    // ---- bias + softmax (rows live in 16-lane groups) ----
    const float* bp = biasHNN + h * 4096;
    bf16* psw = Ps + wave * 16 * 72;
    #pragma unroll
    for (int r = 0; r < 8; ++r) {
        const int qrow = wave * 16 + r + hi * 8;
        float sv[4];
        #pragma unroll
        for (int nt = 0; nt < 4; ++nt)
            sv[nt] = s[nt][r] + bp[qrow * 64 + nt * 16 + lane16];

        float m = fmaxf(fmaxf(sv[0], sv[1]), fmaxf(sv[2], sv[3]));
        #pragma unroll
        for (int off = 8; off >= 1; off >>= 1) m = fmaxf(m, __shfl_xor(m, off, 32));

        float e[4], sum = 0.f;
        #pragma unroll
        for (int nt = 0; nt < 4; ++nt) { e[nt] = __expf(sv[nt] - m); sum += e[nt]; }
        #pragma unroll
        for (int off = 8; off >= 1; off >>= 1) sum += __shfl_xor(sum, off, 32);
        const float inv = 1.f / sum;

        const int rowt = r + hi * 8;
        #pragma unroll
        for (int nt = 0; nt < 4; ++nt)
            psw[rowt * 72 + nt * 16 + lane16] = f2bf(e[nt] * inv);
    }

    asm volatile("s_wait_dscnt 0" ::: "memory");   // P strip visible before fragment reads

    // ---- O = P @ V  (Vs rows are d, cols are tokens -> contiguous B-frags) ----
    v8f o[4] = {zero, zero, zero, zero};
    #pragma unroll
    for (int k0 = 0; k0 < NTOK; k0 += 32) {
        V16 a;
        const bf16* app = psw + lane16 * 72 + k0 + (hi ? 8 : 0);
        a.h[0] = *(const v8bf*)app;
        a.h[1] = *(const v8bf*)(app + 16);
        #pragma unroll
        for (int nt = 0; nt < 4; ++nt) {
            V16 bm;
            const bf16* bvp = Vs + (nt * 16 + lane16) * 72 + k0 + (hi ? 8 : 0);
            bm.h[0] = *(const v8bf*)bvp;
            bm.h[1] = *(const v8bf*)(bvp + 16);
            o[nt] = __builtin_amdgcn_wmma_f32_16x16x32_bf16(false, a.v, false, bm.v,
                                                            (short)0, o[nt], false, false);
        }
    }

    // ---- store context bf16: ctx[(b*64+tok)*768 + h*64 + d] ----
    #pragma unroll
    for (int nt = 0; nt < 4; ++nt) {
        const int d = nt * 16 + lane16;
        #pragma unroll
        for (int r = 0; r < 8; ++r) {
            const int tok = wave * 16 + r + hi * 8;
            ctx[(size_t)(b * 64 + tok) * DIM + h * 64 + d] = f2bf(o[nt][r]);
        }
    }
}

extern "C" void kernel_launch(void* const* d_in, const int* in_sizes, int n_in,
                              void* d_out, int out_size, void* d_ws, size_t ws_size,
                              hipStream_t stream) {
    const float* x      = (const float*)d_in[0];
    const float* qkv_w  = (const float*)d_in[1];
    const float* qkv_b  = (const float*)d_in[2];
    const float* proj_w = (const float*)d_in[3];
    const float* proj_b = (const float*)d_in[4];
    const float* rpb    = (const float*)d_in[5];
    const int*   rpi    = (const int*)d_in[6];
    float* out = (float*)d_out;

    char* ws = (char*)d_ws;
    bf16*  qw      = (bf16*)ws;  ws += (size_t)DIM * 3 * DIM * 2;      // 3.5 MB
    bf16*  pw      = (bf16*)ws;  ws += (size_t)DIM * DIM * 2;          // 1.2 MB
    float* biasHNN = (float*)ws; ws += (size_t)HEADS * 64 * 64 * 4;    // 196 KB
    const size_t qkvBytes = (size_t)1024 * HEADS * 64 * 64 * 2;        // 100.7 MB each
    bf16* Qb  = (bf16*)ws; ws += qkvBytes;
    bf16* Kb  = (bf16*)ws; ws += qkvBytes;
    bf16* Vt  = (bf16*)ws; ws += qkvBytes;
    bf16* ctx = (bf16*)ws; ws += qkvBytes;

    convert_f32_bf16<<<(DIM * 3 * DIM + 255) / 256, 256, 0, stream>>>(qkv_w, qw, DIM * 3 * DIM);
    convert_f32_bf16<<<(DIM * DIM + 255) / 256, 256, 0, stream>>>(proj_w, pw, DIM * DIM);
    bias_gather<<<(HEADS * 4096) / 256, 256, 0, stream>>>(rpb, rpi, biasHNN);

    // QKV projection: (65536 x 768) @ (768 x 2304)
    gemm_k<0><<<dim3(3 * DIM / 64, 65536 / 128), 256, 0, stream>>>(
        x, qw, qkv_b, Qb, Kb, Vt, nullptr, 65536, DIM, 3 * DIM);

    // windowed attention, one block per (window, head)
    attn_k<<<1024 * HEADS, 128, 0, stream>>>(Qb, Kb, Vt, biasHNN, ctx);

    // output projection: (65536 x 768) @ (768 x 768)
    gemm_k<1><<<dim3(DIM / 64, 65536 / 128), 256, 0, stream>>>(
        ctx, pw, proj_b, nullptr, nullptr, nullptr, out, 65536, DIM, DIM);
}
